// RPE_Attention_22943715295341
// MI455X (gfx1250) — compile-verified, matrix operand order checked by reading, not run on hardware
//
#include <hip/hip_runtime.h>
#include <hip/hip_bf16.h>

// ---------------------------------------------------------------------------
// RPE attention for MI455X (gfx1250, wave32, WMMA).
// Round 4: vectorized (v4f) softmax scans over the LDS score rows and the
// normalize pass folded into the P-fragment build (per-row 1/sum applied
// during f32->bf16 conversion) -- one full LDS pass removed.
// GEMMs keep the round-3 structure: 2x2 WMMA tiles, branch-free b128 loads.
// ---------------------------------------------------------------------------

typedef __bf16 bf16;
typedef bf16  v16bf __attribute__((ext_vector_type(16)));
typedef bf16  v8bf  __attribute__((ext_vector_type(8)));
typedef bf16  v2bf  __attribute__((ext_vector_type(2)));
typedef float v8f   __attribute__((ext_vector_type(8)));
typedef float v4f   __attribute__((ext_vector_type(4)));

#define HEADS 12
#define NTOK  197      // 196 patches + CLS
#define NPAD  224      // 14 tiles of 16; 7 * 32 for the P@V K-loop
#define CDIM  768
#define HD    64
#define BATCH 64
#define MROWS (BATCH * NTOK)   // 12608 = 788 * 16 = 394 * 32 exactly

// ----- fp32 -> bf16 conversion helpers -------------------------------------
__device__ __forceinline__ bf16 cvt_bf16(float x) {   // scalar RNE
  unsigned u = __builtin_bit_cast(unsigned, x);
  u += 0x7FFFu + ((u >> 16) & 1u);
  unsigned short h = (unsigned short)(u >> 16);
  return __builtin_bit_cast(bf16, h);
}

__device__ __forceinline__ v2bf cvt2_bf16(float lo, float hi) {
#if __has_builtin(__builtin_amdgcn_cvt_pk_bf16_f32)
  auto t = __builtin_amdgcn_cvt_pk_bf16_f32(lo, hi);
  return __builtin_bit_cast(v2bf, t);
#else
  v2bf r;
  r[0] = cvt_bf16(lo);
  r[1] = cvt_bf16(hi);
  return r;
#endif
}

// A/B fragment: element e holds K = kbase + (e<8 ? e : e+8) + 8*(lane>=16)
// -> exactly two 16B contiguous runs: two b128 loads + shuffle concat.
__device__ __forceinline__ v16bf frag_from_bf16(const bf16* __restrict__ row,
                                                int kbase, int grp) {
  const v8bf lo = *(const v8bf*)(row + kbase + grp * 8);
  const v8bf hi = *(const v8bf*)(row + kbase + grp * 8 + 16);
  return __builtin_shufflevector(lo, hi, 0, 1, 2, 3, 4, 5, 6, 7,
                                         8, 9, 10, 11, 12, 13, 14, 15);
}

__device__ __forceinline__ v8f wmma_bf16(v16bf a, v16bf b, v8f c) {
  // 8 args: (neg_a, A, neg_b, B, c_mod, C, reuse_a, reuse_b)
  return __builtin_amdgcn_wmma_f32_16x16x32_bf16(false, a, false, b,
                                                 (short)0, c, false, false);
}

// ---------------------------------------------------------------------------
// Pre-pass: bulk fp32 -> bf16 (8 elements / thread, b128 in, b128 out)
// ---------------------------------------------------------------------------
__global__ void __launch_bounds__(256) cvt_bulk_kernel(
    const float* __restrict__ src, bf16* __restrict__ dst, int n8) {
  const int i = blockIdx.x * 256 + threadIdx.x;
  if (i >= n8) return;
  const v4f* s = (const v4f*)src + (size_t)i * 2;
  const v4f a = s[0], b = s[1];
  v8bf o;
  v2bf t;
  t = cvt2_bf16(a[0], a[1]); o[0] = t[0]; o[1] = t[1];
  t = cvt2_bf16(a[2], a[3]); o[2] = t[0]; o[3] = t[1];
  t = cvt2_bf16(b[0], b[1]); o[4] = t[0]; o[5] = t[1];
  t = cvt2_bf16(b[2], b[3]); o[6] = t[0]; o[7] = t[1];
  *((v8bf*)dst + i) = o;
}

// ---------------------------------------------------------------------------
// Pre-pass: dense bias[h][q][k] = rpb_table[rel_idx[q*196+k]*12 + h]
// ---------------------------------------------------------------------------
__global__ void __launch_bounds__(256) bias_expand_kernel(
    const float* __restrict__ table, const int* __restrict__ rel_idx,
    float* __restrict__ bias_ws) {
  const int i = blockIdx.x * 256 + threadIdx.x;
  const int total = HEADS * 196 * 196;
  if (i >= total) return;
  const int h = i / (196 * 196);
  const int r = i - h * 196 * 196;
  bias_ws[i] = table[rel_idx[r] * HEADS + h];
}

// ---------------------------------------------------------------------------
// Kernel 1: qkv = xbf @ wqkv_bf^T, 32x32 output per wave (2x2 WMMA tiles).
// Pad token rows (t >= 197) are CLAMPED to row 196 -> finite garbage, which is
// safe: K-pad logits are masked by assignment, V-pad columns meet exact-zero P,
// Q-pad rows are never stored.
//   q_ws/k_ws : [B, H, NPAD, HD] bf16 (q pre-scaled by 1/8)
//   vt_ws     : [B, H, HD, NPAD] bf16 (transposed for P@V B-fragments)
// ---------------------------------------------------------------------------
__global__ void __launch_bounds__(128) qkv_kernel(
    const bf16* __restrict__ xbf, const bf16* __restrict__ wqkv_bf,
    bf16* __restrict__ q_ws, bf16* __restrict__ k_ws, bf16* __restrict__ vt_ws) {
  const int lane = threadIdx.x & 31;
  const int wid  = threadIdx.x >> 5;
  int tid = blockIdx.x * 4 + wid;
  const int ntp = tid % 72; tid /= 72;          // N pair: 2304/32
  const int ttp = tid % 7;                      // M pair: NPAD/32
  const int b   = tid / 7;
  const int grp  = lane >> 4;
  const int lrow = lane & 15;

  const int t0 = ttp * 32 + lrow;               // tile-0 row, tile-1 = +16
  const int t1 = t0 + 16;
  const int c0 = (t0 < NTOK) ? t0 : (NTOK - 1); // clamp: finite, branch-free
  const int c1 = (t1 < NTOK) ? t1 : (NTOK - 1);
  const bf16* arow0 = xbf + ((size_t)b * NTOK + c0) * CDIM;
  const bf16* arow1 = xbf + ((size_t)b * NTOK + c1) * CDIM;
  const bf16* brow0 = wqkv_bf + (size_t)(ntp * 32 + lrow) * CDIM;
  const bf16* brow1 = brow0 + (size_t)16 * CDIM;

  v8f acc00 = {}, acc01 = {}, acc10 = {}, acc11 = {};
  for (int kk = 0; kk < CDIM; kk += 32) {
    __builtin_prefetch(arow0 + kk + 128, 0, 3);
    __builtin_prefetch(brow0 + kk + 128, 0, 3);
    const v16bf a0 = frag_from_bf16(arow0, kk, grp);
    const v16bf a1 = frag_from_bf16(arow1, kk, grp);
    const v16bf b0 = frag_from_bf16(brow0, kk, grp);
    const v16bf b1 = frag_from_bf16(brow1, kk, grp);
    acc00 = wmma_bf16(a0, b0, acc00);
    acc01 = wmma_bf16(a0, b1, acc01);
    acc10 = wmma_bf16(a1, b0, acc10);
    acc11 = wmma_bf16(a1, b1, acc11);
  }

  // decode (uniform across the 32-col pair: 32 | 64 | 768)
  const int which = (ntp * 32) / CDIM;          // 0=q 1=k 2=v
  const int nn0   = ntp * 32 - which * CDIM;
  const int h     = nn0 / HD;
  const int d0    = (nn0 % HD) + lrow;          // tile-0 column
  const int d1    = d0 + 16;                    // tile-1 column (< 64)
  const float scl = (which == 0) ? 0.125f : 1.0f;  // 1/sqrt(64)
  const int rowoff = (lane < 16) ? 0 : 8;
  const size_t bh  = (size_t)b * HEADS + h;

  const v8f accs[4] = {acc00, acc01, acc10, acc11};
  const int mb[4]   = {ttp * 32, ttp * 32, ttp * 32 + 16, ttp * 32 + 16};
  const int dd[4]   = {d0, d1, d0, d1};
#pragma unroll
  for (int q = 0; q < 4; ++q) {
#pragma unroll
    for (int r = 0; r < 8; ++r) {
      const int m = mb[q] + r + rowoff;
      const bf16 val = cvt_bf16(accs[q][r] * scl);
      if (which == 2) {
        vt_ws[(bh * HD + dd[q]) * NPAD + m] = val;
      } else {
        bf16* dst = (which == 0) ? q_ws : k_ws;
        dst[(bh * NPAD + m) * HD + dd[q]] = val;
      }
    }
  }
}

// ---------------------------------------------------------------------------
// Kernel 2: per (b, h, 16-row q-tile) wave:
//   S = QK^T (+bias, key>=197 masked by assignment) -> LDS, softmax in LDS
//   (exp unnormalized; 1/sum folded into P-fragment build), O = P@V,
//   store bf16 to attn_ws[b*197+t][h*64+d].
// ---------------------------------------------------------------------------
__global__ void __launch_bounds__(128) attn_kernel(
    const bf16* __restrict__ q_ws, const bf16* __restrict__ k_ws,
    const bf16* __restrict__ vt_ws, const float* __restrict__ bias_ws,
    bf16* __restrict__ attn_ws) {
  __shared__ float sc[4][16][NPAD];             // 57344 B per 4-wave block
  const int lane = threadIdx.x & 31;
  const int wid  = threadIdx.x >> 5;
  int tid = blockIdx.x * 4 + wid;
  const int qtile = tid % 13; tid /= 13;        // only tiles touching t<197
  const int h     = tid % HEADS;
  const int b     = tid / HEADS;
  const int grp  = lane >> 4;
  const int lrow = lane & 15;
  const int rowoff = (lane < 16) ? 0 : 8;
  float (*S)[NPAD] = sc[wid];
  const size_t bh = (size_t)b * HEADS + h;

  // Q fragments live in VGPRs for the whole tile (K = hd = 64 -> 2 frags)
  const bf16* qrow = q_ws + (bh * NPAD + qtile * 16 + lrow) * HD;
  const v16bf qa0 = frag_from_bf16(qrow, 0, grp);
  const v16bf qa1 = frag_from_bf16(qrow, 32, grp);

  for (int kt = 0; kt < NPAD / 16; ++kt) {
    const bf16* krow = k_ws + (bh * NPAD + kt * 16 + lrow) * HD;
    v8f acc = {};
    acc = wmma_bf16(qa0, frag_from_bf16(krow, 0, grp), acc);
    acc = wmma_bf16(qa1, frag_from_bf16(krow, 32, grp), acc);
    const int ki = kt * 16 + lrow;
#pragma unroll
    for (int r = 0; r < 8; ++r) {
      const int m  = r + rowoff;
      const int qi = qtile * 16 + m;
      float s = acc[r];
      if (ki >= NTOK)
        s = -__builtin_inff();                  // assignment: kills any garbage
      else if (qi >= 1 && qi < NTOK && ki >= 1)
        s += bias_ws[((size_t)h * 196 + (qi - 1)) * 196 + (ki - 1)];
      S[m][ki] = s;
    }
  }

  // softmax stats: lane handles row lrow, half grp (112 cols, v4f scans);
  // partners combined via shfl_xor 16. exp stays unnormalized in LDS.
  float inv;
  {
    const int cbase = grp * 112;
    float mx = -__builtin_inff();
    for (int c = 0; c < 112; c += 4) {
      const v4f v = *(const v4f*)&S[lrow][cbase + c];
      mx = fmaxf(mx, fmaxf(fmaxf(v[0], v[1]), fmaxf(v[2], v[3])));
    }
    mx = fmaxf(mx, __shfl_xor(mx, 16, 32));
    float sum = 0.f;
    for (int c = 0; c < 112; c += 4) {
      v4f v = *(v4f*)&S[lrow][cbase + c];
      v4f e;
      e[0] = __expf(v[0] - mx);
      e[1] = __expf(v[1] - mx);
      e[2] = __expf(v[2] - mx);
      e[3] = __expf(v[3] - mx);
      *(v4f*)&S[lrow][cbase + c] = e;
      sum += (e[0] + e[1]) + (e[2] + e[3]);
    }
    sum += __shfl_xor(sum, 16, 32);
    inv = 1.0f / sum;                           // same for both row partners
  }

  // O = P @ V : 4 d-tiles of 16, K over 224 keys (masked keys are exact 0);
  // per-row 1/sum applied during f32->bf16 fragment conversion.
  v8f o0 = {}, o1 = {}, o2 = {}, o3 = {};
  for (int kk = 0; kk < NPAD; kk += 32) {
    v16bf pa;
    {
      const v4f* p0 = (const v4f*)&S[lrow][kk + grp * 8];
      const v4f* p1 = (const v4f*)&S[lrow][kk + grp * 8 + 16];
      const v4f a0 = p0[0], a1 = p0[1], b0 = p1[0], b1 = p1[1];
      v2bf t;
      t = cvt2_bf16(a0[0] * inv, a0[1] * inv); pa[0]  = t[0]; pa[1]  = t[1];
      t = cvt2_bf16(a0[2] * inv, a0[3] * inv); pa[2]  = t[0]; pa[3]  = t[1];
      t = cvt2_bf16(a1[0] * inv, a1[1] * inv); pa[4]  = t[0]; pa[5]  = t[1];
      t = cvt2_bf16(a1[2] * inv, a1[3] * inv); pa[6]  = t[0]; pa[7]  = t[1];
      t = cvt2_bf16(b0[0] * inv, b0[1] * inv); pa[8]  = t[0]; pa[9]  = t[1];
      t = cvt2_bf16(b0[2] * inv, b0[3] * inv); pa[10] = t[0]; pa[11] = t[1];
      t = cvt2_bf16(b1[0] * inv, b1[1] * inv); pa[12] = t[0]; pa[13] = t[1];
      t = cvt2_bf16(b1[2] * inv, b1[3] * inv); pa[14] = t[0]; pa[15] = t[1];
    }
    const bf16* vbase = vt_ws + (bh * HD + lrow) * NPAD;
    o0 = wmma_bf16(pa, frag_from_bf16(vbase + 0 * NPAD * 16, kk, grp), o0);
    o1 = wmma_bf16(pa, frag_from_bf16(vbase + 1 * NPAD * 16, kk, grp), o1);
    o2 = wmma_bf16(pa, frag_from_bf16(vbase + 2 * NPAD * 16, kk, grp), o2);
    o3 = wmma_bf16(pa, frag_from_bf16(vbase + 3 * NPAD * 16, kk, grp), o3);
  }

  const v8f oacc[4] = {o0, o1, o2, o3};
#pragma unroll
  for (int dt = 0; dt < 4; ++dt) {
#pragma unroll
    for (int r = 0; r < 8; ++r) {
      const int tq = qtile * 16 + r + rowoff;
      if (tq < NTOK)
        attn_ws[((size_t)b * NTOK + tq) * CDIM + h * HD + dt * 16 + lrow] =
            cvt_bf16(oacc[dt][r]);
    }
  }
}

// ---------------------------------------------------------------------------
// Kernel 3: out = attn_ws @ wproj_bf^T + b_proj, 32x32 per wave.
// M = 12608 = 394 * 32 exactly; N = 768 = 24 * 32 exactly.
// ---------------------------------------------------------------------------
__global__ void __launch_bounds__(128) proj_kernel(
    const bf16* __restrict__ attn_ws, const bf16* __restrict__ wproj_bf,
    const float* __restrict__ b_proj, float* __restrict__ out) {
  const int lane = threadIdx.x & 31;
  const int wid  = threadIdx.x >> 5;
  int tid = blockIdx.x * 4 + wid;
  const int ntp = tid % 24;                     // 768/32
  const int mtp = tid / 24;                     // 394 pairs
  const int grp = lane >> 4, lrow = lane & 15;

  const bf16* arow0 = attn_ws + (size_t)(mtp * 32 + lrow) * CDIM;
  const bf16* arow1 = arow0 + (size_t)16 * CDIM;
  const bf16* brow0 = wproj_bf + (size_t)(ntp * 32 + lrow) * CDIM;
  const bf16* brow1 = brow0 + (size_t)16 * CDIM;

  v8f acc00 = {}, acc01 = {}, acc10 = {}, acc11 = {};
  for (int kk = 0; kk < CDIM; kk += 32) {
    __builtin_prefetch(arow0 + kk + 128, 0, 3);
    __builtin_prefetch(brow0 + kk + 128, 0, 3);
    const v16bf a0 = frag_from_bf16(arow0, kk, grp);
    const v16bf a1 = frag_from_bf16(arow1, kk, grp);
    const v16bf b0 = frag_from_bf16(brow0, kk, grp);
    const v16bf b1 = frag_from_bf16(brow1, kk, grp);
    acc00 = wmma_bf16(a0, b0, acc00);
    acc01 = wmma_bf16(a0, b1, acc01);
    acc10 = wmma_bf16(a1, b0, acc10);
    acc11 = wmma_bf16(a1, b1, acc11);
  }

  const int rowoff = (lane < 16) ? 0 : 8;
  const int n0 = ntp * 32 + lrow;
  const int n1 = n0 + 16;
  const float bias0 = b_proj[n0];
  const float bias1 = b_proj[n1];

  const v8f accs[4]  = {acc00, acc01, acc10, acc11};
  const int mb[4]    = {mtp * 32, mtp * 32, mtp * 32 + 16, mtp * 32 + 16};
  const int nn[4]    = {n0, n1, n0, n1};
  const float bs[4]  = {bias0, bias1, bias0, bias1};
#pragma unroll
  for (int q = 0; q < 4; ++q) {
#pragma unroll
    for (int r = 0; r < 8; ++r) {
      const int m = mb[q] + r + rowoff;
      out[(size_t)m * CDIM + nn[q]] = accs[q][r] + bs[q];
    }
  }
}

// ---------------------------------------------------------------------------
extern "C" void kernel_launch(void* const* d_in, const int* in_sizes, int n_in,
                              void* d_out, int out_size, void* d_ws, size_t ws_size,
                              hipStream_t stream) {
  (void)in_sizes; (void)n_in; (void)out_size; (void)ws_size;
  const float* x      = (const float*)d_in[0];  // [64,197,768]
  const float* w_qkv  = (const float*)d_in[1];  // [2304,768]
  const float* w_proj = (const float*)d_in[2];  // [768,768]
  const float* b_proj = (const float*)d_in[3];  // [768]
  const float* rpb    = (const float*)d_in[4];  // [729,12]
  const int*   relidx = (const int*)d_in[5];    // [196,196]
  float* out = (float*)d_out;

  // workspace carve (all regions 256B aligned)
  const size_t XBF_N   = (size_t)MROWS * CDIM;          // 9,682,944
  const size_t WQKV_N  = (size_t)3 * CDIM * CDIM;       // 1,769,472
  const size_t WPROJ_N = (size_t)CDIM * CDIM;           //   589,824
  const size_t QKV_BYTES  = (size_t)BATCH * HEADS * NPAD * HD * sizeof(bf16);
  const size_t BIAS_BYTES = (size_t)HEADS * 196 * 196 * sizeof(float);
  const size_t ATTN_BYTES = XBF_N * sizeof(bf16);
  auto align256 = [](size_t v) { return (v + 255) & ~(size_t)255; };
  char* ws = (char*)d_ws;
  bf16*  xbf_ws   = (bf16*)ws;  ws += align256(XBF_N * sizeof(bf16));
  bf16*  wqkv_ws  = (bf16*)ws;  ws += align256(WQKV_N * sizeof(bf16));
  bf16*  wproj_ws = (bf16*)ws;  ws += align256(WPROJ_N * sizeof(bf16));
  bf16*  q_ws     = (bf16*)ws;  ws += align256(QKV_BYTES);
  bf16*  k_ws     = (bf16*)ws;  ws += align256(QKV_BYTES);
  bf16*  vt_ws    = (bf16*)ws;  ws += align256(QKV_BYTES);
  float* bias_ws  = (float*)ws; ws += align256(BIAS_BYTES);
  bf16*  attn_ws  = (bf16*)ws;  ws += align256(ATTN_BYTES);

  // one-shot conversions + bias gather (independent; serialized on stream)
  cvt_bulk_kernel<<<(int)((XBF_N / 8 + 255) / 256), 256, 0, stream>>>(
      x, xbf_ws, (int)(XBF_N / 8));
  cvt_bulk_kernel<<<(int)((WQKV_N / 8 + 255) / 256), 256, 0, stream>>>(
      w_qkv, wqkv_ws, (int)(WQKV_N / 8));
  cvt_bulk_kernel<<<(int)((WPROJ_N / 8 + 255) / 256), 256, 0, stream>>>(
      w_proj, wproj_ws, (int)(WPROJ_N / 8));
  bias_expand_kernel<<<(HEADS * 196 * 196 + 255) / 256, 256, 0, stream>>>(
      rpb, relidx, bias_ws);

  // qkv GEMM — 64 * 7 * 72 waves (32x32 tiles), 4 waves/block
  qkv_kernel<<<(BATCH * 7 * 72) / 4, 128, 0, stream>>>(
      xbf_ws, wqkv_ws, q_ws, k_ws, vt_ws);

  // attention — 64 * 12 * 13 waves, 4 waves/block
  attn_kernel<<<(BATCH * HEADS * 13) / 4, 128, 0, stream>>>(
      q_ws, k_ws, vt_ws, bias_ws, attn_ws);

  // output projection — 394 * 24 waves (32x32 tiles), 4 waves/block
  proj_kernel<<<(394 * 24) / 4, 128, 0, stream>>>(
      attn_ws, wproj_ws, b_proj, out);
}
